// ACAM_24876450579320
// MI455X (gfx1250) — compile-verified, matrix operand order checked by reading, not run on hardware
//
#include <hip/hip_runtime.h>

// ---------------------------------------------------------------------------
// ACAM fused inference for MI455X (gfx1250, wave32, WMMA 16x16x32 f16)
// ---------------------------------------------------------------------------

typedef __attribute__((ext_vector_type(16))) _Float16 v16h;
typedef __attribute__((ext_vector_type(8)))  _Float16 v8h;
typedef __attribute__((ext_vector_type(8)))  float    v8f;

#define BSZ   32768
#define WW    7
#define FF    80
#define WF    560
#define WFP   576      // K padded to multiple of 32 for ei1
#define HH    512
#define H4    2048
#define NPASS 8        // initial pass + 7 scan steps
#define EPSV  1e-5f

static __device__ __forceinline__ v8f vzero8() {
  v8f r;
#pragma unroll
  for (int i = 0; i < 8; ++i) r[i] = 0.0f;
  return r;
}

// Load a 16x32 f16 A-fragment from LDS (row-major, `stride` halves per row)
// per ISA 7.12.2: lane L: M = L&15; halves[0..7]=K(kk+kh+0..7),
// halves[8..15]=K(kk+16+kh+0..7), kh = (L>=16)?8:0.
static __device__ __forceinline__ v16h load_a_lds(const _Float16* base, int stride, int kk) {
  const int lane = threadIdx.x & 31;
  const int m    = lane & 15;
  const int kh   = (lane >> 4) << 3;
  const v8h x0 = *(const v8h*)(base + m * stride + kk + kh);
  const v8h x1 = *(const v8h*)(base + m * stride + kk + 16 + kh);
  v16h r;
#pragma unroll
  for (int i = 0; i < 8; ++i) { r[i] = x0[i]; r[i + 8] = x1[i]; }
  return r;
}

// Load a pre-packed 32x16 f16 B-fragment from global memory.
// Packing: block of 512 halves = 32 lanes * 16 halves, lane-major.
static __device__ __forceinline__ v16h load_b_pk(const _Float16* blk) {
  const int lane = threadIdx.x & 31;
  const v8h x0 = *(const v8h*)(blk + lane * 16);
  const v8h x1 = *(const v8h*)(blk + lane * 16 + 8);
  v16h r;
#pragma unroll
  for (int i = 0; i < 8; ++i) { r[i] = x0[i]; r[i + 8] = x1[i]; }
  return r;
}

static __device__ __forceinline__ v8f wmma_f16(v16h a, v16h b, v8f c) {
  return __builtin_amdgcn_wmma_f32_16x16x32_f16(false, a, false, b, (short)0, c, false, false);
}

static __device__ __forceinline__ float sigm(float x) { return 1.0f / (1.0f + __expf(-x)); }

// ---------------------------------------------------------------------------
// Prep kernels
// ---------------------------------------------------------------------------

// BN fold: s = g*rsqrt(v+eps); b' = (b-m)*s + beta
__global__ void k_fold_bn(const float* __restrict__ b, const float* __restrict__ g,
                          const float* __restrict__ beta, const float* __restrict__ m,
                          const float* __restrict__ v, float* __restrict__ s_out,
                          float* __restrict__ b_out, int n) {
  int i = blockIdx.x * blockDim.x + threadIdx.x;
  if (i >= n) return;
  float s = g[i] * rsqrtf(v[i] + EPSV);
  s_out[i] = s;
  b_out[i] = (b[i] - m[i]) * s + beta[i];
}

// Scale rows of a row-major f32 weight (used for VALU-consumed ea1 / dec)
__global__ void k_scale_rows(const float* __restrict__ w, const float* __restrict__ s,
                             float* __restrict__ out, int nOut, int nIn) {
  int i = blockIdx.x * blockDim.x + threadIdx.x;
  if (i >= nOut * nIn) return;
  int o = i / nIn;
  out[i] = w[i] * s[o];
}

// Pack W[out][in] (optionally BN-row-scaled) into WMMA B-fragment blocks.
// Packed index p: h=p&15, lane=(p>>4)&31, ks=(p>>9)%ksteps, nt=p/(512*ksteps)
__global__ void k_pack_b(const float* __restrict__ w, const float* __restrict__ s,
                         _Float16* __restrict__ out, int nOut, int nIn, int ksteps) {
  int total = (nOut >> 4) * ksteps * 512;
  int p = blockIdx.x * blockDim.x + threadIdx.x;
  if (p >= total) return;
  int h    = p & 15;
  int lane = (p >> 4) & 31;
  int ks   = (p >> 9) % ksteps;
  int nt   = p / (ksteps << 9);
  int n    = (nt << 4) + (lane & 15);
  int kb   = (ks << 5) + ((lane >> 4) << 3) + ((h < 8) ? h : (16 + (h - 8)));
  float val = 0.0f;
  if (kb < nIn) {
    val = w[n * nIn + kb];
    if (s) val *= s[n];
  }
  out[p] = (_Float16)val;
}

__global__ void k_bias_sum(const float* __restrict__ a, const float* __restrict__ b,
                           float* __restrict__ out, int n) {
  int i = blockIdx.x * blockDim.x + threadIdx.x;
  if (i < n) out[i] = a[i] + b[i];
}

// ---------------------------------------------------------------------------
// Main fused kernel: one block per 16 batch rows, 8 waves, whole recurrence
// ---------------------------------------------------------------------------
__global__ __launch_bounds__(256, 1) void k_acam_main(
    const float* __restrict__ features,
    const _Float16* __restrict__ wihp, const _Float16* __restrict__ whhp,
    const _Float16* __restrict__ ea2p, const _Float16* __restrict__ ei1p,
    const _Float16* __restrict__ ei2p,
    const float* __restrict__ ea1fw, const float* __restrict__ ea1fb,
    const float* __restrict__ ea2fb, const float* __restrict__ ei1fb,
    const float* __restrict__ ei2fb,
    const float* __restrict__ decfw, const float* __restrict__ decfb,
    const float* __restrict__ bsum,
    const float* __restrict__ ln_g, const float* __restrict__ ln_b,
    const float* __restrict__ cls_w, const float* __restrict__ cls_b,
    float* __restrict__ out) {
  __shared__ __align__(16) _Float16 sX[16 * WFP];   // staging: t1 / ai / t2 / agg (A-frags)
  __shared__ __align__(16) _Float16 sH[16 * HH];    // ln(h) A-frags
  __shared__ __align__(16) _Float16 sHraw[16 * HH]; // raw h (dec / classifier / ln source)
  __shared__ float sAtt[16 * 8];
  __shared__ float sRed[8 * 16 * 4];
  __shared__ float sStats[16 * 4];                  // per row: mu_h, inv_h, mu_c, inv_c
  __shared__ float sOut[16 * 14];

  const int tid  = threadIdx.x;
  const int lane = tid & 31;
  const int wave = tid >> 5;
  const int g0   = blockIdx.x * 16;
  const int ncol = (lane & 15);          // N within a 16-wide tile (C/D layout)
  const int mrow0 = (lane >> 4) << 3;    // row base for this lane's C/D VGPRs

  // Warm L2/WGP$ with this block's features tile (re-read every pass).
  {
    const char* fb = (const char*)(features + (size_t)g0 * WF);
    for (int off = tid * 128; off < 16 * WF * 4; off += 256 * 128)
      __builtin_prefetch(fb + off, 0, 1);
  }

  // Zero sX once: pad columns 560..575 are never written again, so every
  // per-pass pad guard in the staging loops disappears.
  for (int p = tid; p < 16 * WFP; p += 256) sX[p] = (_Float16)0.0f;
  for (int p = tid; p < 16 * HH; p += 256) {
    sH[p]    = (_Float16)0.0f;
    sHraw[p] = (_Float16)0.0f;
  }

  float c_reg[4][8];                     // c state in D-matrix layout, 4 N-tiles/wave
#pragma unroll
  for (int t = 0; t < 4; ++t)
#pragma unroll
    for (int i = 0; i < 8; ++i) c_reg[t][i] = 0.0f;

  for (int pass = 0; pass < NPASS; ++pass) {
    // ---- 1. attention weights -> sAtt[r][j] (normalized) ----
    if (pass == 0) {
      for (int p = tid; p < 16 * 8; p += 256) sAtt[p] = 1.0f / 7.0f;
      __syncthreads();
    } else {
      if (tid < 112) {
        int r = tid / 7, j = tid % 7;
        float acc = decfb[j];
        const _Float16* hp = &sHraw[r * HH];
        const float* wp = &decfw[j * HH];
        for (int k = 0; k < HH; ++k) acc += (float)hp[k] * wp[k];
        sAtt[r * 8 + j] = sigm(acc);
      }
      __syncthreads();
      if (tid < 16) {
        float s = 0.0f;
#pragma unroll
        for (int j = 0; j < 7; ++j) s += sAtt[tid * 8 + j];
        float inv = 1.0f / s;
#pragma unroll
        for (int j = 0; j < 7; ++j) sAtt[tid * 8 + j] *= inv;
      }
      __syncthreads();
    }

    // ---- 2. t1 = relu(BNfold(ea1) @ att)  (K=7, VALU) -> sX f16 ----
    for (int p = tid; p < 16 * HH; p += 256) {
      int r = p >> 9, o = p & 511;
      float acc = ea1fb[o];
      const float* wp = &ea1fw[o * 7];
      const float* ap = &sAtt[r * 8];
#pragma unroll
      for (int j = 0; j < 7; ++j) acc += ap[j] * wp[j];
      sX[r * WFP + o] = (_Float16)fmaxf(acc, 0.0f);
    }
    __syncthreads();

    // ---- 3. a = BNfold(ea2) @ t1  (WMMA, K=512), kept in regs ----
    // K-outer: one shared A-fragment feeds 4 independent accumulator chains.
    v8f a_acc[4];
#pragma unroll
    for (int t = 0; t < 4; ++t) a_acc[t] = vzero8();
#pragma clang loop unroll(disable)
    for (int ks = 0; ks < 16; ++ks) {
      v16h A = load_a_lds(sX, WFP, ks << 5);
#pragma unroll
      for (int t = 0; t < 4; ++t) {
        int nt = wave * 4 + t;
        a_acc[t] = wmma_f16(A, load_b_pk(ea2p + ((size_t)(nt * 16 + ks) << 9)), a_acc[t]);
      }
    }
#pragma unroll
    for (int t = 0; t < 4; ++t) {
      float bb = ea2fb[(wave * 4 + t) * 16 + ncol];
#pragma unroll
      for (int i = 0; i < 8; ++i) a_acc[t][i] += bb;
    }
    __syncthreads();  // everyone done reading t1 in sX

    // ---- 4. ai = (features * att).flatten -> sX f16 (pad cols stay zero) ----
    for (int p = tid; p < 16 * WF; p += 256) {
      int r = p / WF;
      int k = p - r * WF;
      sX[r * WFP + k] =
          (_Float16)(features[(size_t)(g0 + r) * WF + k] * sAtt[r * 8 + (k / FF)]);
    }
    __syncthreads();

    // ---- 5. t2 = relu(BNfold(ei1) @ ai)  (WMMA, K=576) ----
    v8f t_acc[4];
#pragma unroll
    for (int t = 0; t < 4; ++t) t_acc[t] = vzero8();
#pragma clang loop unroll(disable)
    for (int ks = 0; ks < 18; ++ks) {
      v16h A = load_a_lds(sX, WFP, ks << 5);
#pragma unroll
      for (int t = 0; t < 4; ++t) {
        int nt = wave * 4 + t;
        t_acc[t] = wmma_f16(A, load_b_pk(ei1p + ((size_t)(nt * 18 + ks) << 9)), t_acc[t]);
      }
    }
#pragma unroll
    for (int t = 0; t < 4; ++t) {
      float bb = ei1fb[(wave * 4 + t) * 16 + ncol];
#pragma unroll
      for (int i = 0; i < 8; ++i) t_acc[t][i] = fmaxf(t_acc[t][i] + bb, 0.0f);
    }
    __syncthreads();  // done reading ai
#pragma unroll
    for (int t = 0; t < 4; ++t) {
      int col = (wave * 4 + t) * 16 + ncol;
#pragma unroll
      for (int i = 0; i < 8; ++i) sX[(mrow0 + i) * WFP + col] = (_Float16)t_acc[t][i];
    }
    __syncthreads();

    // ---- 6. x = BNfold(ei2) @ t2; agg = relu(a + x) -> sX f16 ----
    v8f x_acc[4];
#pragma unroll
    for (int t = 0; t < 4; ++t) x_acc[t] = vzero8();
#pragma clang loop unroll(disable)
    for (int ks = 0; ks < 16; ++ks) {
      v16h A = load_a_lds(sX, WFP, ks << 5);
#pragma unroll
      for (int t = 0; t < 4; ++t) {
        int nt = wave * 4 + t;
        x_acc[t] = wmma_f16(A, load_b_pk(ei2p + ((size_t)(nt * 16 + ks) << 9)), x_acc[t]);
      }
    }
#pragma unroll
    for (int t = 0; t < 4; ++t) {
      float bb = ei2fb[(wave * 4 + t) * 16 + ncol];
#pragma unroll
      for (int i = 0; i < 8; ++i)
        x_acc[t][i] = fmaxf(x_acc[t][i] + bb + a_acc[t][i], 0.0f);
    }
    __syncthreads();  // done reading t2
#pragma unroll
    for (int t = 0; t < 4; ++t) {
      int col = (wave * 4 + t) * 16 + ncol;
#pragma unroll
      for (int i = 0; i < 8; ++i) sX[(mrow0 + i) * WFP + col] = (_Float16)x_acc[t][i];
    }

    // ---- 7. ln(h) -> sH (stats precomputed at previous gate update) ----
    if (pass > 0) {
      for (int p = tid; p < 16 * HH; p += 256) {
        int r = p >> 9, k2 = p & 511;
        float mu = sStats[r * 4 + 0], inv = sStats[r * 4 + 1];
        float hv = (float)sHraw[p];
        sH[p] = (_Float16)(((hv - mu) * inv) * ln_g[k2] + ln_b[k2]);
      }
    }
    __syncthreads();  // agg + sH ready for gate WMMAs

    // ---- 8. LSTM gates: i,f,g,o = agg@Wih^T + ln(h)@Whh^T + (b_ih+b_hh) ----
    float hs[8], hq[8], cs[8], cq[8];
#pragma unroll
    for (int i = 0; i < 8; ++i) { hs[i] = hq[i] = cs[i] = cq[i] = 0.0f; }

#pragma unroll
    for (int t = 0; t < 4; ++t) {
      int nt = wave * 4 + t;
      v8f gi = vzero8(), gf = vzero8(), gg = vzero8(), go = vzero8();
      // 8 independent load->WMMA chains per iteration: keep the loop rolled so
      // the register allocator never hoists 16 iterations of B-fragments.
#pragma clang loop unroll(disable)
      for (int ks = 0; ks < 16; ++ks) {
        v16h Ax = load_a_lds(sX, WFP, ks << 5);
        v16h Ah = load_a_lds(sH, HH, ks << 5);
        gi = wmma_f16(Ax, load_b_pk(wihp + ((size_t)((0 * 32 + nt) * 16 + ks) << 9)), gi);
        gf = wmma_f16(Ax, load_b_pk(wihp + ((size_t)((1 * 32 + nt) * 16 + ks) << 9)), gf);
        gg = wmma_f16(Ax, load_b_pk(wihp + ((size_t)((2 * 32 + nt) * 16 + ks) << 9)), gg);
        go = wmma_f16(Ax, load_b_pk(wihp + ((size_t)((3 * 32 + nt) * 16 + ks) << 9)), go);
        gi = wmma_f16(Ah, load_b_pk(whhp + ((size_t)((0 * 32 + nt) * 16 + ks) << 9)), gi);
        gf = wmma_f16(Ah, load_b_pk(whhp + ((size_t)((1 * 32 + nt) * 16 + ks) << 9)), gf);
        gg = wmma_f16(Ah, load_b_pk(whhp + ((size_t)((2 * 32 + nt) * 16 + ks) << 9)), gg);
        go = wmma_f16(Ah, load_b_pk(whhp + ((size_t)((3 * 32 + nt) * 16 + ks) << 9)), go);
      }
      int col = nt * 16 + ncol;
      float bi = bsum[0 * HH + col], bf = bsum[1 * HH + col];
      float bg = bsum[2 * HH + col], bo = bsum[3 * HH + col];
      float lg = ln_g[col], lb = ln_b[col];
#pragma unroll
      for (int i = 0; i < 8; ++i) {
        int m = mrow0 + i;
        float cin = c_reg[t][i];
        if (pass > 0) {
          float mu = sStats[m * 4 + 2], inv = sStats[m * 4 + 3];
          cin = (cin - mu) * inv * lg + lb;
        }
        float iv = sigm(gi[i] + bi);
        float fv = sigm(gf[i] + bf);
        float gv = tanhf(gg[i] + bg);
        float ov = sigm(go[i] + bo);
        float cn = fv * cin + iv * gv;
        float hn = ov * tanhf(cn);
        c_reg[t][i] = cn;
        sHraw[m * HH + col] = (_Float16)hn;
        hs[i] += hn; hq[i] += hn * hn; cs[i] += cn; cq[i] += cn * cn;
      }
    }

    // ---- 9. in-register LN stats (wave32 butterflies within 16-lane halves) ----
#pragma unroll
    for (int mask = 1; mask < 16; mask <<= 1) {
#pragma unroll
      for (int i = 0; i < 8; ++i) {
        hs[i] += __shfl_xor(hs[i], mask, 32);
        hq[i] += __shfl_xor(hq[i], mask, 32);
        cs[i] += __shfl_xor(cs[i], mask, 32);
        cq[i] += __shfl_xor(cq[i], mask, 32);
      }
    }
    if ((lane & 15) == 0) {
#pragma unroll
      for (int i = 0; i < 8; ++i) {
        int r = mrow0 + i;
        sRed[(wave * 16 + r) * 4 + 0] = hs[i];
        sRed[(wave * 16 + r) * 4 + 1] = hq[i];
        sRed[(wave * 16 + r) * 4 + 2] = cs[i];
        sRed[(wave * 16 + r) * 4 + 3] = cq[i];
      }
    }
    __syncthreads();
    if (tid < 16) {
      float sh = 0, sh2 = 0, sc = 0, sc2 = 0;
#pragma unroll
      for (int w = 0; w < 8; ++w) {
        sh  += sRed[(w * 16 + tid) * 4 + 0];
        sh2 += sRed[(w * 16 + tid) * 4 + 1];
        sc  += sRed[(w * 16 + tid) * 4 + 2];
        sc2 += sRed[(w * 16 + tid) * 4 + 3];
      }
      float muh = sh * (1.0f / 512.0f);
      float muc = sc * (1.0f / 512.0f);
      sStats[tid * 4 + 0] = muh;
      sStats[tid * 4 + 1] = rsqrtf(sh2 * (1.0f / 512.0f) - muh * muh + EPSV);
      sStats[tid * 4 + 2] = muc;
      sStats[tid * 4 + 3] = rsqrtf(sc2 * (1.0f / 512.0f) - muc * muc + EPSV);
    }
    __syncthreads();
  }  // passes

  // ---- classifier + log_softmax over (7,2) pairs ----
  if (tid < 224) {
    int r = tid / 14, j = tid % 14;
    float acc = cls_b[j];
    const _Float16* hp = &sHraw[r * HH];
    const float* wp = &cls_w[j * HH];
    for (int k = 0; k < HH; ++k) acc += (float)hp[k] * wp[k];
    sOut[r * 14 + j] = acc;
  }
  __syncthreads();
  if (tid < 224) {
    int r = tid / 14, j = tid % 14, p = j >> 1;
    float a0 = sOut[r * 14 + p * 2], a1 = sOut[r * 14 + p * 2 + 1];
    float mx = fmaxf(a0, a1);
    float lse = mx + __logf(__expf(a0 - mx) + __expf(a1 - mx));
    out[((size_t)(g0 + r) * 7 + p) * 2 + (j & 1)] = sOut[r * 14 + j] - lse;
  }
}

// ---------------------------------------------------------------------------
// Host launch
// ---------------------------------------------------------------------------
extern "C" void kernel_launch(void* const* d_in, const int* in_sizes, int n_in,
                              void* d_out, int out_size, void* d_ws, size_t ws_size,
                              hipStream_t stream) {
  const float* features = (const float*)d_in[0];
  const float* dec_w = (const float*)d_in[1];
  const float* dec_b = (const float*)d_in[2];
  const float* dec_g = (const float*)d_in[3];
  const float* dec_bt = (const float*)d_in[4];
  const float* dec_m = (const float*)d_in[5];
  const float* dec_v = (const float*)d_in[6];
  const float* ea1_w = (const float*)d_in[7];
  const float* ea1_b = (const float*)d_in[8];
  const float* ea1_g = (const float*)d_in[9];
  const float* ea1_bt = (const float*)d_in[10];
  const float* ea1_m = (const float*)d_in[11];
  const float* ea1_v = (const float*)d_in[12];
  const float* ea2_w = (const float*)d_in[13];
  const float* ea2_b = (const float*)d_in[14];
  const float* ea2_g = (const float*)d_in[15];
  const float* ea2_bt = (const float*)d_in[16];
  const float* ea2_m = (const float*)d_in[17];
  const float* ea2_v = (const float*)d_in[18];
  const float* ei1_w = (const float*)d_in[19];
  const float* ei1_b = (const float*)d_in[20];
  const float* ei1_g = (const float*)d_in[21];
  const float* ei1_bt = (const float*)d_in[22];
  const float* ei1_m = (const float*)d_in[23];
  const float* ei1_v = (const float*)d_in[24];
  const float* ei2_w = (const float*)d_in[25];
  const float* ei2_b = (const float*)d_in[26];
  const float* ei2_g = (const float*)d_in[27];
  const float* ei2_bt = (const float*)d_in[28];
  const float* ei2_m = (const float*)d_in[29];
  const float* ei2_v = (const float*)d_in[30];
  const float* w_ih = (const float*)d_in[31];
  const float* w_hh = (const float*)d_in[32];
  const float* b_ih = (const float*)d_in[33];
  const float* b_hh = (const float*)d_in[34];
  const float* ln_g = (const float*)d_in[35];
  const float* ln_b = (const float*)d_in[36];
  const float* cls_w = (const float*)d_in[37];
  const float* cls_b = (const float*)d_in[38];

  // Workspace carve-up (256B aligned)
  char* ws = (char*)d_ws;
  size_t off = 0;
  auto carve = [&](size_t bytes) -> char* {
    char* p = ws + off;
    off = (off + bytes + 255) & ~(size_t)255;
    return p;
  };
  _Float16* wihp = (_Float16*)carve((size_t)H4 * HH * 2);
  _Float16* whhp = (_Float16*)carve((size_t)H4 * HH * 2);
  _Float16* ea2p = (_Float16*)carve((size_t)HH * HH * 2);
  _Float16* ei1p = (_Float16*)carve((size_t)HH * WFP * 2);
  _Float16* ei2p = (_Float16*)carve((size_t)HH * HH * 2);
  float* ea1fw = (float*)carve((size_t)HH * 7 * 4);
  float* ea1fb = (float*)carve(HH * 4);
  float* ea1s  = (float*)carve(HH * 4);
  float* ea2fb = (float*)carve(HH * 4);
  float* ea2s  = (float*)carve(HH * 4);
  float* ei1fb = (float*)carve(HH * 4);
  float* ei1s  = (float*)carve(HH * 4);
  float* ei2fb = (float*)carve(HH * 4);
  float* ei2s  = (float*)carve(HH * 4);
  float* decfw = (float*)carve((size_t)7 * HH * 4);
  float* decfb = (float*)carve(32);
  float* decs  = (float*)carve(32);
  float* bsum  = (float*)carve(H4 * 4);
  (void)ws_size; (void)n_in; (void)in_sizes; (void)out_size;

  // --- BN folding ---
  k_fold_bn<<<2, 256, 0, stream>>>(ea1_b, ea1_g, ea1_bt, ea1_m, ea1_v, ea1s, ea1fb, HH);
  k_fold_bn<<<2, 256, 0, stream>>>(ea2_b, ea2_g, ea2_bt, ea2_m, ea2_v, ea2s, ea2fb, HH);
  k_fold_bn<<<2, 256, 0, stream>>>(ei1_b, ei1_g, ei1_bt, ei1_m, ei1_v, ei1s, ei1fb, HH);
  k_fold_bn<<<2, 256, 0, stream>>>(ei2_b, ei2_g, ei2_bt, ei2_m, ei2_v, ei2s, ei2fb, HH);
  k_fold_bn<<<1, 32, 0, stream>>>(dec_b, dec_g, dec_bt, dec_m, dec_v, decs, decfb, 7);

  // --- VALU-consumed f32 folded weights ---
  k_scale_rows<<<(HH * 7 + 255) / 256, 256, 0, stream>>>(ea1_w, ea1s, ea1fw, HH, 7);
  k_scale_rows<<<(7 * HH + 255) / 256, 256, 0, stream>>>(dec_w, decs, decfw, 7, HH);

  // --- WMMA B-fragment packing (BN folded where applicable) ---
  k_pack_b<<<(HH * HH + 255) / 256, 256, 0, stream>>>(ea2_w, ea2s, ea2p, HH, HH, 16);
  k_pack_b<<<(HH * WFP + 255) / 256, 256, 0, stream>>>(ei1_w, ei1s, ei1p, HH, WF, 18);
  k_pack_b<<<(HH * HH + 255) / 256, 256, 0, stream>>>(ei2_w, ei2s, ei2p, HH, HH, 16);
  k_pack_b<<<((size_t)H4 * HH + 255) / 256, 256, 0, stream>>>(w_ih, nullptr, wihp, H4, HH, 16);
  k_pack_b<<<((size_t)H4 * HH + 255) / 256, 256, 0, stream>>>(w_hh, nullptr, whhp, H4, HH, 16);
  k_bias_sum<<<(H4 + 255) / 256, 256, 0, stream>>>(b_ih, b_hh, bsum, H4);

  // --- fused recurrent kernel: 2048 blocks x 256 threads (8 wave32 waves) ---
  k_acam_main<<<BSZ / 16, 256, 0, stream>>>(
      features, wihp, whhp, ea2p, ei1p, ei2p,
      ea1fw, ea1fb, ea2fb, ei1fb, ei2fb,
      decfw, decfb, bsum, ln_g, ln_b, cls_w, cls_b,
      (float*)d_out);
}
